// MahalanobisBlock_27324581937365
// MI455X (gfx1250) — matrix operand order repaired and checked by reading, not compiled, not run on hardware
//
#include <hip/hip_runtime.h>
#include <hip/hip_bf16.h>
#include <stdint.h>

// ---------------------------------------------------------------------------
// Mahalanobis similarity for MI455X (gfx1250, wave32, WMMA).
//   k_means  : per-class channel means                 [J,C]
//   k_cov    : exact fp32 sample covariance + REG*I    [J,C,C]
//   k_invert : register-resident Gauss-Jordan inverse, emits bf16 hi/lo split
//              pre-swizzled into the WMMA 16-bit A-operand lane layout
//   k_query  : per-(b,c) L2-normalize + center, store transposed [b][p][c]
//   k_mahal  : fused diag(diff^T inv diff) via v_wmma_f32_16x16x32_bf16,
//              3-term hi/lo split (AhBh + AhBl + AlBh ~ fp32 accuracy).
//              512 threads = 16 waves, one 16-row c-tile per wave, 7 16-col
//              N-tiles per workgroup: one A fetch (64 B/lane) feeds 21 WMMAs.
//              No outer reuse loop exists, so B LDS loads cannot be LICM'd
//              into registers (round-2 spill bug); ks loop is unroll-1 and
//              sched_barriers keep B live ranges short.
// ---------------------------------------------------------------------------

typedef __attribute__((ext_vector_type(16))) __bf16 v16bf;
typedef __attribute__((ext_vector_type(8)))  float  v8f;

#define J_CLASSES 5
#define BS        5
#define C_DIM     256
#define P_DIM     441
#define N_SAMP    (BS * P_DIM)   /* 2205 */
#define B_QUERY   128
#define REG_EPS   1e-6f

#define NTILES    7              /* 16-wide N tiles per workgroup -> 112 p  */
#define TILE_P    (NTILES * 16)
#define PBLOCKS   ((P_DIM + TILE_P - 1) / TILE_P)   /* 4 */
#define NWAVES    16             /* one c-tile per wave                     */

union V16BF {
    uint4 q[2];
    v16bf v;
};

// ------------------------------- kernel 1: class means ---------------------
__global__ void k_means(const float* __restrict__ x2, float* __restrict__ mean) {
    int idx = blockIdx.x;            // j*C + c
    int j = idx >> 8;
    int c = idx & 255;
    float s = 0.f;
    for (int bs = 0; bs < BS; ++bs) {
        const float* row = x2 + ((size_t)(j * BS + bs) * C_DIM + c) * P_DIM;
        for (int p = threadIdx.x; p < P_DIM; p += 128) s += row[p];
    }
    __shared__ float red[128];
    red[threadIdx.x] = s;
    __syncthreads();
    for (int off = 64; off > 0; off >>= 1) {
        if ((int)threadIdx.x < off) red[threadIdx.x] += red[threadIdx.x + off];
        __syncthreads();
    }
    if (threadIdx.x == 0) mean[idx] = red[0] / (float)N_SAMP;
}

// ------------------------------- kernel 2: covariance (exact fp32) ---------
__global__ void k_cov(const float* __restrict__ x2, const float* __restrict__ mean,
                      float* __restrict__ cov) {
    int j  = blockIdx.z;
    int ca = blockIdx.x * 32;
    int cb = blockIdx.y * 32;
    int tx = threadIdx.x & 31;       // n within chunk (coalesced p)
    int ty = threadIdx.x >> 5;       // 0..7 (wave-uniform)
    __shared__ float As[32][33];
    __shared__ float Bt[32][33];
    float acc[4] = {0.f, 0.f, 0.f, 0.f};
    for (int n0 = 0; n0 < N_SAMP; n0 += 32) {
        int n  = n0 + tx;
        int bs = n / P_DIM;
        int p  = n - bs * P_DIM;
        bool ok = n < N_SAMP;
        #pragma unroll
        for (int rr = 0; rr < 4; ++rr) {
            int cl = ty + 8 * rr;
            float a = 0.f, b = 0.f;
            if (ok) {
                a = x2[((size_t)(j * BS + bs) * C_DIM + ca + cl) * P_DIM + p] - mean[j * C_DIM + ca + cl];
                b = x2[((size_t)(j * BS + bs) * C_DIM + cb + cl) * P_DIM + p] - mean[j * C_DIM + cb + cl];
            }
            As[tx][cl] = a;
            Bt[tx][cl] = b;
        }
        __syncthreads();
        #pragma unroll 8
        for (int kk = 0; kk < 32; ++kk) {
            float bv = Bt[kk][tx];
            #pragma unroll
            for (int r = 0; r < 4; ++r)
                acc[r] = fmaf(As[kk][ty + 8 * r], bv, acc[r]);
        }
        __syncthreads();
    }
    #pragma unroll
    for (int r = 0; r < 4; ++r) {
        int c1 = ca + ty + 8 * r;
        int c2 = cb + tx;
        float v = acc[r] / (float)(N_SAMP - 1);
        if (c1 == c2) v += REG_EPS;
        cov[((size_t)j * C_DIM + c1) * C_DIM + c2] = v;
    }
}

// ------------------------------- kernel 3: Gauss-Jordan inverse ------------
__global__ void __launch_bounds__(1024) k_invert(float* __restrict__ cov,
        unsigned short* __restrict__ invH, unsigned short* __restrict__ invL) {
    int j   = blockIdx.x;
    int tid = threadIdx.x;
    int cchunk = tid >> 8;           // 0..3, wave-uniform
    int row    = tid & 255;
    float* M = cov + (size_t)j * C_DIM * C_DIM;
    float m[64];
    #pragma unroll
    for (int jj = 0; jj < 64; ++jj) m[jj] = M[(size_t)row * C_DIM + cchunk * 64 + jj];

    __shared__ float rowbuf[256];
    __shared__ float colbuf[256];

    for (int k = 0; k < C_DIM; ++k) {
        int kc = k >> 6;
        int kk = k & 63;
        if (cchunk == kc) {                    // wave-uniform branch
            float msel = m[0];
            #pragma unroll
            for (int jj = 1; jj < 64; ++jj) if (jj == kk) msel = m[jj];
            colbuf[row] = msel;
        }
        __syncthreads();
        float pinv = 1.0f / colbuf[k];
        if (row == k) {
            #pragma unroll
            for (int jj = 0; jj < 64; ++jj) rowbuf[cchunk * 64 + jj] = m[jj] * pinv;
        }
        __syncthreads();
        float f = colbuf[row];
        if (row == k) {
            #pragma unroll
            for (int jj = 0; jj < 64; ++jj) m[jj] = rowbuf[cchunk * 64 + jj];
            if (cchunk == kc) {
                #pragma unroll
                for (int jj = 0; jj < 64; ++jj) if (jj == kk) m[jj] = pinv;
            }
        } else {
            #pragma unroll
            for (int jj = 0; jj < 64; ++jj) m[jj] = fmaf(-f, rowbuf[cchunk * 64 + jj], m[jj]);
            if (cchunk == kc) {
                float fp = -f * pinv;
                #pragma unroll
                for (int jj = 0; jj < 64; ++jj) if (jj == kk) m[jj] = fp;
            }
        }
        __syncthreads();
    }

    // write swizzled bf16 hi/lo, WMMA 16-bit A-operand layout (ISA 7.12.2)
    int c = row;
    int ctile = c >> 4, mrow = c & 15;
    #pragma unroll
    for (int jj = 0; jj < 64; ++jj) {
        int d    = cchunk * 64 + jj;
        int ks   = d >> 5;
        int k2   = d & 31;
        int half = (k2 >> 3) & 1;
        int kp   = k2 - 8 * half;                                // {0..7} U {16..23}
        int v    = (kp >= 16) ? (4 + ((kp - 16) >> 1)) : (kp >> 1);
        int slot = kp & 1;
        int lane = mrow + 16 * half;
        size_t idx = ((((size_t)j * 16 + ctile) * 8 + ks) * 32 + lane) * 16 + v * 2 + slot;
        float val = m[jj];
        __bf16 h = (__bf16)val;
        __bf16 l = (__bf16)(val - (float)h);
        invH[idx] = __builtin_bit_cast(unsigned short, h);
        invL[idx] = __builtin_bit_cast(unsigned short, l);
    }
}

// ------------------------------- kernel 4: query normalize/center ----------
__global__ void k_query(const float* __restrict__ x1, float* __restrict__ dTF) {
    int idx = blockIdx.x;            // b*C + c
    int b = idx >> 8;
    int c = idx & 255;
    const float* q = x1 + (size_t)idx * P_DIM;
    float s1 = 0.f, s2 = 0.f;
    for (int p = threadIdx.x; p < P_DIM; p += 128) {
        float v = q[p];
        s1 += v;
        s2 += v * v;
    }
    __shared__ float r1[128], r2[128];
    r1[threadIdx.x] = s1; r2[threadIdx.x] = s2;
    __syncthreads();
    for (int off = 64; off > 0; off >>= 1) {
        if ((int)threadIdx.x < off) {
            r1[threadIdx.x] += r1[threadIdx.x + off];
            r2[threadIdx.x] += r2[threadIdx.x + off];
        }
        __syncthreads();
    }
    float scale = 1.0f / sqrtf(r2[0]);
    float sub   = r1[0] / (float)P_DIM;
    for (int p = threadIdx.x; p < P_DIM; p += 128) {
        float dv = (q[p] - sub) * scale;
        dTF[((size_t)b * P_DIM + p) * C_DIM + c] = dv;   // transposed [b][p][c]
    }
}

// ------------------------------- kernel 5: fused quadratic form ------------
__global__ void __launch_bounds__(512) k_mahal(const float* __restrict__ dTF,
        const unsigned short* __restrict__ invH, const unsigned short* __restrict__ invL,
        float* __restrict__ out) {
    int p0  = blockIdx.x * TILE_P;
    int j   = blockIdx.y;
    int b   = blockIdx.z;
    int tid = threadIdx.x;

    extern __shared__ __align__(16) char smem[];
    uint32_t* BH  = (uint32_t*)smem;                          // 8*7*32*8 dwords
    uint32_t* BL  = BH + 8 * NTILES * 32 * 8;
    float*    red = (float*)(BL + 8 * NTILES * 32 * 8);       // [16][7][32]

    // ---- stage B operand: diff[b, d, p-block] -> bf16 hi/lo, B layout ----
    for (int idx = tid; idx < 8 * NTILES * 32; idx += 512) {
        int lane = idx & 31;
        int t    = idx >> 5;
        int nt   = t % NTILES;
        int ks   = t / NTILES;
        int hf   = lane >> 4;
        int p    = p0 + nt * 16 + (lane & 15);
        int d0   = ks * 32 + hf * 8;
        uint32_t hw[8], lw[8];
        if (p < P_DIM) {
            const float* src = dTF + ((size_t)b * P_DIM + p) * C_DIM + d0; // two runs of 8 f32: d0..d0+7, d0+16..d0+23
            #pragma unroll
            for (int v = 0; v < 8; ++v) {
                float f0 = src[2 * v + (v >= 4 ? 8 : 0)], f1 = src[2 * v + 1 + (v >= 4 ? 8 : 0)];
                __bf16 h0 = (__bf16)f0, h1 = (__bf16)f1;
                __bf16 l0 = (__bf16)(f0 - (float)h0), l1 = (__bf16)(f1 - (float)h1);
                hw[v] = (uint32_t)__builtin_bit_cast(unsigned short, h0) |
                        ((uint32_t)__builtin_bit_cast(unsigned short, h1) << 16);
                lw[v] = (uint32_t)__builtin_bit_cast(unsigned short, l0) |
                        ((uint32_t)__builtin_bit_cast(unsigned short, l1) << 16);
            }
        } else {
            #pragma unroll
            for (int v = 0; v < 8; ++v) { hw[v] = 0u; lw[v] = 0u; }
        }
        #pragma unroll
        for (int v = 0; v < 8; ++v) {
            BH[((ks * NTILES + nt) * 32 + lane) * 8 + v] = hw[v];
            BL[((ks * NTILES + nt) * 32 + lane) * 8 + v] = lw[v];
        }
    }
    __syncthreads();

    int ct   = __builtin_amdgcn_readfirstlane(tid >> 5);     // scalar: wave id == c-tile
    int lane = tid & 31;
    int half = lane >> 4;

    v8f acc[NTILES];
    #pragma unroll
    for (int nt = 0; nt < NTILES; ++nt)
        acc[nt] = (v8f){0.f, 0.f, 0.f, 0.f, 0.f, 0.f, 0.f, 0.f};

    const uint4* iH4 = (const uint4*)invH;
    const uint4* iL4 = (const uint4*)invL;

    #pragma unroll 1
    for (int ks = 0; ks < 8; ++ks) {
        size_t abase = ((((size_t)j * 16 + ct) * 8 + ks) * 32 + lane) * 2;  // uint4 units
        V16BF Ah, Al;
        Ah.q[0] = iH4[abase]; Ah.q[1] = iH4[abase + 1];
        Al.q[0] = iL4[abase]; Al.q[1] = iL4[abase + 1];
        #pragma unroll
        for (int nt = 0; nt < NTILES; ++nt) {
            V16BF Bh, Bl;
            const uint4* bh4 = (const uint4*)&BH[((ks * NTILES + nt) * 32 + lane) * 8];
            const uint4* bl4 = (const uint4*)&BL[((ks * NTILES + nt) * 32 + lane) * 8];
            Bh.q[0] = bh4[0]; Bh.q[1] = bh4[1];
            Bl.q[0] = bl4[0]; Bl.q[1] = bl4[1];
            acc[nt] = __builtin_amdgcn_wmma_f32_16x16x32_bf16(false, Ah.v, false, Bh.v, (short)0, acc[nt], false, false);
            acc[nt] = __builtin_amdgcn_wmma_f32_16x16x32_bf16(false, Ah.v, false, Bl.v, (short)0, acc[nt], false, false);
            acc[nt] = __builtin_amdgcn_wmma_f32_16x16x32_bf16(false, Al.v, false, Bh.v, (short)0, acc[nt], false, false);
            __builtin_amdgcn_sched_barrier(0);   // keep B live ranges short (no spills)
        }
    }

    // acc[nt][r] = T[ct*16 + r + 8*half][p_nt]  (C/D layout: VGPR r -> M=r / M=r+8)
    #pragma unroll
    for (int nt = 0; nt < NTILES; ++nt) {
        float partial = 0.f;
        int p = p0 + nt * 16 + (lane & 15);
        if (p < P_DIM) {
            const float* df = dTF + ((size_t)b * P_DIM + p) * C_DIM + ct * 16 + half * 8;
            #pragma unroll
            for (int r = 0; r < 8; ++r)
                partial = fmaf(acc[nt][r], df[r], partial);
        }
        red[(ct * NTILES + nt) * 32 + lane] = partial;
    }
    __syncthreads();

    if (tid < NTILES * 16) {
        int nt = tid >> 4;
        int pi = tid & 15;
        float s = 0.f;
        #pragma unroll
        for (int w = 0; w < NWAVES; ++w)
            s += red[(w * NTILES + nt) * 32 + pi] + red[(w * NTILES + nt) * 32 + pi + 16];
        int pp = p0 + nt * 16 + pi;
        if (pp < P_DIM) out[(size_t)b * (J_CLASSES * P_DIM) + j * P_DIM + pp] = s;
    }
}

// ------------------------------- launcher ----------------------------------
extern "C" void kernel_launch(void* const* d_in, const int* in_sizes, int n_in,
                              void* d_out, int out_size, void* d_ws, size_t ws_size,
                              hipStream_t stream) {
    (void)in_sizes; (void)n_in; (void)out_size; (void)ws_size;
    const float* x1 = (const float*)d_in[0];   // [128,256,21,21]
    const float* x2 = (const float*)d_in[1];   // [5,5,256,21,21]
    float* out = (float*)d_out;                // [128, 5*441]

    char* ws = (char*)d_ws;
    size_t off = 0;
    auto alloc = [&](size_t bytes) {
        size_t o = off;
        off = (off + bytes + 255) & ~(size_t)255;
        return o;
    };
    float*          mean = (float*)(ws + alloc((size_t)J_CLASSES * C_DIM * 4));
    float*          cov  = (float*)(ws + alloc((size_t)J_CLASSES * C_DIM * C_DIM * 4));
    unsigned short* invH = (unsigned short*)(ws + alloc((size_t)J_CLASSES * C_DIM * C_DIM * 2));
    unsigned short* invL = (unsigned short*)(ws + alloc((size_t)J_CLASSES * C_DIM * C_DIM * 2));
    float*          dTF  = (float*)(ws + alloc((size_t)B_QUERY * P_DIM * C_DIM * 4));
    // total workspace ~60.5 MB

    k_means <<<J_CLASSES * C_DIM, 128, 0, stream>>>(x2, mean);
    k_cov   <<<dim3(8, 8, J_CLASSES), 256, 0, stream>>>(x2, mean, cov);
    k_invert<<<J_CLASSES, 1024, 0, stream>>>(cov, invH, invL);
    k_query <<<B_QUERY * C_DIM, 128, 0, stream>>>(x1, dTF);

    size_t lds_bytes = (size_t)(8 * NTILES * 32 * 8) * 4 * 2       // BH + BL
                     + (size_t)(NWAVES * NTILES * 32) * 4;         // red
    k_mahal <<<dim3(PBLOCKS, J_CLASSES, B_QUERY), 512, lds_bytes, stream>>>(dTF, invH, invL, out);
}
